// Model_84086869721456
// MI455X (gfx1250) — matrix operand order repaired and checked by reading, not compiled, not run on hardware
//
#include <hip/hip_runtime.h>
#include <hip/hip_bf16.h>
#include <math.h>
#include <stdint.h>

// ---------------------------------------------------------------------------
// Types for CDNA5 WMMA (wave32):  D(16x16 f32) = A(16x32 bf16) x B(32x16 bf16) + C
// ---------------------------------------------------------------------------
typedef __attribute__((ext_vector_type(16))) __bf16 v16bf;
typedef __attribute__((ext_vector_type(8)))  float  v8f;
typedef __attribute__((ext_vector_type(4)))  float  v4f;
// Exact pointee type of the async-LDS builtin per hipcc's diagnostic:
typedef int v4i_b128 __attribute__((vector_size(4 * sizeof(int))));

#define BM 64
#define BN 128
#define BK 32

// ---------------------------------------------------------------------------
// Async global->LDS staging (CDNA5). Prefer builtin, else inline asm, else copy.
// ---------------------------------------------------------------------------
#if __has_builtin(__builtin_amdgcn_global_load_async_to_lds_b128)
#define ASYNC_MODE 2
#elif defined(__gfx1250__)
#define ASYNC_MODE 1
#else
#define ASYNC_MODE 0
#endif

__device__ __forceinline__ void cp16_async(const float* g, float* l) {
#if ASYNC_MODE == 2
  __builtin_amdgcn_global_load_async_to_lds_b128(
      (__attribute__((address_space(1))) v4i_b128*)(uintptr_t)g,
      (__attribute__((address_space(3))) v4i_b128*)(uint32_t)(uintptr_t)l,
      0, 0);
#elif ASYNC_MODE == 1
  asm volatile("global_load_async_to_lds_b128 %0, %1, off"
               :: "v"((uint32_t)(uintptr_t)l), "v"((uint64_t)(uintptr_t)g)
               : "memory");
#else
  *(v4f*)l = *(const v4f*)g;
#endif
}

__device__ __forceinline__ void wait_async() {
#if ASYNC_MODE == 2
#if __has_builtin(__builtin_amdgcn_s_wait_asynccnt)
  __builtin_amdgcn_s_wait_asynccnt(0);
#else
  asm volatile("s_wait_asynccnt 0x0" ::: "memory");
#endif
#elif ASYNC_MODE == 1
  asm volatile("s_wait_asynccnt 0x0" ::: "memory");
#endif
}

__device__ __forceinline__ __bf16 f2bf(float f) {
#if defined(__gfx1250__)
  // Native fptrunc: lowers to v_cvt_pk_bf16_f32 (RNE) on CDNA5, pairs get packed.
  return (__bf16)f;
#else
  unsigned u = __float_as_uint(f);
  unsigned r = u + 0x7FFFu + ((u >> 16) & 1u);   // round-to-nearest-even
  unsigned short h = (unsigned short)(r >> 16);
  __bf16 b;
  __builtin_memcpy(&b, &h, 2);
  return b;
#endif
}

__device__ __forceinline__ float gelu_exact(float x) {
  return 0.5f * x * (1.0f + erff(x * 0.70710678118654752440f));
}
__device__ __forceinline__ float sigmoid_f(float x) {
  return 1.0f / (1.0f + __expf(-x));
}

// Build a bf16 fragment from two runs of 8 consecutive f32 in LDS.
__device__ __forceinline__ v16bf make_frag(const float* p0, const float* p1, int aop) {
  v4f a = *(const v4f*)p0, b = *(const v4f*)(p0 + 4);
  v4f c = *(const v4f*)p1, d = *(const v4f*)(p1 + 4);
  v16bf r;
  #pragma unroll
  for (int t = 0; t < 4; ++t) {
    float v0 = a[t], v1 = b[t], v2 = c[t], v3 = d[t];
    if (aop) { v0 = gelu_exact(v0); v1 = gelu_exact(v1);
               v2 = gelu_exact(v2); v3 = gelu_exact(v3); }
    r[t] = f2bf(v0); r[4 + t] = f2bf(v1); r[8 + t] = f2bf(v2); r[12 + t] = f2bf(v3);
  }
  return r;
}

// ---------------------------------------------------------------------------
// Generic WMMA GEMM:  C[M,N] = epi( aop(A[gather]) @ B + bias )
//   REQUIRES K % 32 == 0 (true for all call sites: K in {128,256}).
//   aop    : 0 = identity, 1 = exact gelu on A elements
//   btrans : 0 -> B is [K,N] row-major ; 1 -> B is [N,K] row-major (B^T GEMM)
//   epi    : 0 none | 1 relu | 2 sigmoid | 3 gate-residual
// ---------------------------------------------------------------------------
__global__ __launch_bounds__(256)
void gemm_wmma_bf16(const float* __restrict__ A, long lda,
                    const int* __restrict__ arows, int aop,
                    const float* __restrict__ B, long ldb, int btrans,
                    const float* __restrict__ bias,
                    float* __restrict__ C, long ldc,
                    int M, int N, int K,
                    int epi, const float* __restrict__ gatep,
                    const float* __restrict__ res, long res_ld)
{
  __shared__ __align__(16) float Asf[BM][BK];   // [m][k]  8 KB (f32; async-staged)
  __shared__ __align__(16) float Bsf[BN][BK];   // [n][k] 16 KB (f32; B^T layout)

  const int tid  = threadIdx.x;
  const int lane = tid & 31;
  const int wave = tid >> 5;
  const int wm   = wave & 1;          // 2 waves along M  (2*32 = 64)
  const int wn   = wave >> 1;         // 4 waves along N  (4*32 = 128)
  const int mTile = blockIdx.y * BM;
  const int nTile = blockIdx.x * BN;
  const int hi = (lane >> 4) & 1;     // lane half select
  const int ln = lane & 15;

  v8f acc[2][2];
  #pragma unroll
  for (int i = 0; i < 2; ++i)
    #pragma unroll
    for (int j = 0; j < 2; ++j)
      #pragma unroll
      for (int t = 0; t < 8; ++t) acc[i][j][t] = 0.0f;

  // ---- per-thread staging coordinates (index-clamped: no divergent guards) ----
  const int ar  = tid >> 2;            // A row 0..63
  const int ak0 = (tid & 3) * 8;       // A k-offset {0,8,16,24}
  const int gmA  = mTile + ar;
  const int gmAc = gmA < M ? gmA : (M - 1);
  const long arowA = arows ? (long)arows[gmAc] : (long)gmAc;
  const float* aBase = A + arowA * lda;

  // B (btrans=1) coords
  const int bn1  = tid >> 1;           // 0..127
  const int bk1  = (tid & 1) * 16;
  const int gn1  = nTile + bn1;
  const int gn1c = gn1 < N ? gn1 : (N - 1);
  const float* bBase1 = B + (long)gn1c * ldb;
  // B (btrans=0) coords
  const int bk0 = tid & 31;            // k 0..31
  const int bn0 = (tid >> 5) * 16;     // n group
  const bool fullN = (nTile + BN) <= N;

  for (int kBlk = 0; kBlk < K; kBlk += BK) {
    // ---- stage A tile: 2x async b128 per thread ----
    {
      const float* ap = aBase + kBlk + ak0;
      float* lp = &Asf[ar][ak0];
      cp16_async(ap, lp);
      cp16_async(ap + 4, lp + 4);
    }
    // ---- stage B tile into Bsf[n][k] ----
    if (btrans) {                       // B is [N,K]: contiguous along k -> async
      const float* bp = bBase1 + kBlk + bk1;
      float* lp = &Bsf[bn1][bk1];
      cp16_async(bp,      lp);
      cp16_async(bp + 4,  lp + 4);
      cp16_async(bp + 8,  lp + 8);
      cp16_async(bp + 12, lp + 12);
    } else {                            // B is [K,N]: transpose in VGPRs
      const float* bp = B + (long)(kBlk + bk0) * ldb + nTile + bn0;
      if (fullN) {
        #pragma unroll
        for (int j = 0; j < 16; ++j) Bsf[bn0 + j][bk0] = bp[j];
      } else {
        #pragma unroll
        for (int j = 0; j < 16; ++j) {
          int gn = nTile + bn0 + j;
          Bsf[bn0 + j][bk0] = (gn < N) ? bp[j] : 0.0f;
        }
      }
    }
    wait_async();
    __syncthreads();

    // ---- fragments (f32 LDS -> bf16 regs) + WMMA ----
    const int akb = hi ? 8 : 0;         // ISA A layout: half-lanes take K+8; elems 8..15 at K+16
    v16bf afrag[2];
    #pragma unroll
    for (int si = 0; si < 2; ++si) {
      const float* p = &Asf[wm * 32 + si * 16 + ln][akb];
      afrag[si] = make_frag(p, p + 16, aop);
    }
    const int bkb = hi ? 8 : 0;         // B layout matches A: half-lanes take K+8; elems 8..15 at K+16
    v16bf bfrag[2];
    #pragma unroll
    for (int sj = 0; sj < 2; ++sj) {
      const float* p = &Bsf[wn * 32 + sj * 16 + ln][bkb];
      bfrag[sj] = make_frag(p, p + 16, 0);
    }

    #pragma unroll
    for (int si = 0; si < 2; ++si)
      #pragma unroll
      for (int sj = 0; sj < 2; ++sj)
        acc[si][sj] = __builtin_amdgcn_wmma_f32_16x16x32_bf16(
            false, afrag[si], false, bfrag[sj],
            (short)0, acc[si][sj], false, false);

    __syncthreads();
  }

  // ---- epilogue: bias + activation / gate-residual (C layout: N=ln, M=r+8*hi) ----
  float g = 0.0f;
  if (epi == 3) g = sigmoid_f(gatep[0]);
  #pragma unroll
  for (int si = 0; si < 2; ++si)
    #pragma unroll
    for (int sj = 0; sj < 2; ++sj) {
      int gn = nTile + wn * 32 + sj * 16 + ln;
      if (gn >= N) continue;
      float bv = bias ? bias[gn] : 0.0f;
      #pragma unroll
      for (int r = 0; r < 8; ++r) {
        int gm = mTile + wm * 32 + si * 16 + r + 8 * hi;
        if (gm >= M) continue;
        float c = acc[si][sj][r] + bv;
        if (epi == 1)      c = fmaxf(c, 0.0f);
        else if (epi == 2) c = sigmoid_f(c);
        else if (epi == 3) c = g * c + (1.0f - g) * res[(long)gm * res_ld + gn];
        C[(long)gm * ldc + gn] = c;
      }
    }
}

// ---------------------------------------------------------------------------
// Edge phase (edge_b only: src=paper, dst=author), H=2, D=128, wave32
// ---------------------------------------------------------------------------
__device__ __forceinline__ void atomicMaxF(float* addr, float v) {
  if (v >= 0.0f) atomicMax((int*)addr, __float_as_int(v));
  else           atomicMin((unsigned int*)addr, __float_as_uint(v));
}

__global__ void init_bufs(float* mbuf, long nm, float* sbuf, long ns,
                          float* agg, long na) {
  long i = (long)blockIdx.x * blockDim.x + threadIdx.x;
  long stride = (long)gridDim.x * blockDim.x;
  for (long j = i; j < na; j += stride) agg[j]  = 0.0f;
  for (long j = i; j < nm; j += stride) mbuf[j] = -INFINITY;
  for (long j = i; j < ns; j += stride) sbuf[j] = 0.0f;
}

// One wave per edge: logits[e,h] = (q[dst,h,:] . ke[src,h,:]) * p[h]/sqrt(D)
__global__ void edge_logits(const int* __restrict__ src, const int* __restrict__ dst,
                            const float* __restrict__ q, const float* __restrict__ ke,
                            const float* __restrict__ p, float* __restrict__ logits,
                            float* __restrict__ mbuf, int E) {
  int e = (int)(((long)blockIdx.x * blockDim.x + threadIdx.x) >> 5);
  int lane = threadIdx.x & 31;
  if (e >= E) return;
  int s = src[e], d = dst[e];
  #pragma unroll
  for (int h = 0; h < 2; ++h) {
    const float* qp = q  + (long)d * 256 + h * 128;
    const float* kp = ke + (long)s * 256 + h * 128;
    float part = 0.0f;
    #pragma unroll
    for (int j = 0; j < 4; ++j) part += qp[lane + 32 * j] * kp[lane + 32 * j];
    part += __shfl_xor(part, 16);
    part += __shfl_xor(part, 8);
    part += __shfl_xor(part, 4);
    part += __shfl_xor(part, 2);
    part += __shfl_xor(part, 1);
    if (lane == 0) {
      float lg = part * p[h] * 0.08838834764831845f;  // 1/sqrt(128)
      logits[(long)e * 2 + h] = lg;
      atomicMaxF(mbuf + (long)d * 2 + h, lg);
    }
  }
}

// e = exp(logit - m[dst]); overwrite logits with e; s[dst] += e
__global__ void edge_expsum(const int* __restrict__ dst,
                            float* __restrict__ logits,
                            const float* __restrict__ mbuf,
                            float* __restrict__ sbuf, long EH) {
  long i = (long)blockIdx.x * blockDim.x + threadIdx.x;
  if (i >= EH) return;
  long e = i >> 1; int h = (int)(i & 1);
  int d = dst[e];
  float m = mbuf[(long)d * 2 + h];
  float mm = (m > -1e37f && m < 1e37f) ? m : 0.0f;   // isfinite(m) ? m : 0
  float ex = __expf(logits[i] - mm);
  logits[i] = ex;
  atomicAdd(&sbuf[(long)d * 2 + h], ex);
}

// agg[dst] += (e / (s[dst]+eps)) * ve[src]  (one wave per edge, 256 floats)
__global__ void edge_scatter(const int* __restrict__ src, const int* __restrict__ dst,
                             const float* __restrict__ logits,
                             const float* __restrict__ sbuf,
                             const float* __restrict__ ve,
                             float* __restrict__ agg, int E) {
  int e = (int)(((long)blockIdx.x * blockDim.x + threadIdx.x) >> 5);
  int lane = threadIdx.x & 31;
  if (e >= E) return;
  int s = src[e], d = dst[e];
  #pragma unroll
  for (int h = 0; h < 2; ++h) {
    float a = logits[(long)e * 2 + h] / (sbuf[(long)d * 2 + h] + 1e-16f);
    const float* vp = ve + (long)s * 256 + h * 128;
    float* ag = agg + (long)d * 256 + h * 128;
    #pragma unroll
    for (int j = 0; j < 4; ++j)
      atomicAdd(ag + lane + 32 * j, a * vp[lane + 32 * j]);
  }
}

// ---------------------------------------------------------------------------
// Host-side orchestration
// ---------------------------------------------------------------------------
static inline void launch_gemm(hipStream_t stream,
                               const float* A, long lda, const int* arows, int aop,
                               const float* B, long ldb, int btrans,
                               const float* bias, float* C, long ldc,
                               int M, int N, int K,
                               int epi, const float* gate, const float* res, long res_ld) {
  dim3 grid((unsigned)((N + BN - 1) / BN), (unsigned)((M + BM - 1) / BM));
  gemm_wmma_bf16<<<grid, 256, 0, stream>>>(A, lda, arows, aop, B, ldb, btrans,
                                           bias, C, ldc, M, N, K, epi, gate, res, res_ld);
}

extern "C" void kernel_launch(void* const* d_in, const int* in_sizes, int n_in,
                              void* d_out, int out_size, void* d_ws, size_t ws_size,
                              hipStream_t stream) {
  (void)n_in; (void)out_size; (void)ws_size;
  const float* x_author   = (const float*)d_in[0];
  const float* x_paper    = (const float*)d_in[1];
  const int*   eb_src     = (const int*)d_in[4];
  const int*   eb_dst     = (const int*)d_in[5];
  const int*   input_nodes= (const int*)d_in[6];
  const float* W_in_a     = (const float*)d_in[7];
  const float* b_in_a     = (const float*)d_in[8];
  const float* W_in_p     = (const float*)d_in[9];
  const float* b_in_p     = (const float*)d_in[10];
  const float* Wkqv_a     = (const float*)d_in[11];
  const float* bkqv_a     = (const float*)d_in[12];
  const float* Wkqv_p     = (const float*)d_in[13];
  const float* bkqv_p     = (const float*)d_in[14];
  const float* Wk_rel_b   = (const float*)d_in[17];
  const float* Wv_rel_b   = (const float*)d_in[18];
  const float* p_b        = (const float*)d_in[20];
  const float* Wout_a     = (const float*)d_in[21];
  const float* bout_a     = (const float*)d_in[22];
  const float* skip_a     = (const float*)d_in[25];
  const float* W_dec      = (const float*)d_in[27];
  const float* b_dec      = (const float*)d_in[28];

  const int Na = in_sizes[0] / 128;
  const int Np = in_sizes[1] / 128;
  const int E  = in_sizes[4];
  const int NI = in_sizes[6];
  const int Nmax = Na > Np ? Na : Np;
  const size_t NODE = (size_t)Nmax * 256;

  float* ws      = (float*)d_ws;
  float* xa      = ws;               // xa (live to the end as residual)
  float* xp_qa   = ws + 1 * NODE;    // xp, then q_a
  float* kp_ve   = ws + 2 * NODE;    // k_p, then ve
  float* vp_agg  = ws + 3 * NODE;    // v_p, then agg
  float* ke_za   = ws + 4 * NODE;    // ke,  then za
  float* logits  = ws + 5 * NODE;                   // E*2
  float* mbuf    = logits + (size_t)E * 2;          // Na*2
  float* sbuf    = mbuf   + (size_t)Na * 2;         // Na*2
  float* zin     = sbuf   + (size_t)Na * 2;         // NI*256

  // 1) input projections (relu)
  launch_gemm(stream, x_author, 128, nullptr, 0, W_in_a, 256, 0, b_in_a,
              xa, 256, Na, 256, 128, 1, nullptr, nullptr, 0);
  launch_gemm(stream, x_paper, 128, nullptr, 0, W_in_p, 256, 0, b_in_p,
              xp_qa, 256, Np, 256, 128, 1, nullptr, nullptr, 0);

  // 2) live kqv slices only: k_p (cols 0:256), v_p (cols 512:768), q_a (cols 256:512)
  launch_gemm(stream, xp_qa, 256, nullptr, 0, Wkqv_p + 0, 768, 0, bkqv_p + 0,
              kp_ve, 256, Np, 256, 256, 0, nullptr, nullptr, 0);
  launch_gemm(stream, xp_qa, 256, nullptr, 0, Wkqv_p + 512, 768, 0, bkqv_p + 512,
              vp_agg, 256, Np, 256, 256, 0, nullptr, nullptr, 0);
  launch_gemm(stream, xa, 256, nullptr, 0, Wkqv_a + 256, 768, 0, bkqv_a + 256,
              xp_qa, 256, Na, 256, 256, 0, nullptr, nullptr, 0);   // q_a overwrites xp

  // 3) relation transforms (per head): ke = k_p @ Wk_rel_b[h]
  for (int h = 0; h < 2; ++h)
    launch_gemm(stream, kp_ve + h * 128, 256, nullptr, 0,
                Wk_rel_b + (size_t)h * 128 * 128, 128, 0, nullptr,
                ke_za + h * 128, 256, Np, 128, 128, 0, nullptr, nullptr, 0);
  // ve = v_p @ Wv_rel_b[h]  (over k_p, now dead)
  float* vebuf = kp_ve;
  for (int h = 0; h < 2; ++h)
    launch_gemm(stream, vp_agg + h * 128, 256, nullptr, 0,
                Wv_rel_b + (size_t)h * 128 * 128, 128, 0, nullptr,
                vebuf + h * 128, 256, Np, 128, 128, 0, nullptr, nullptr, 0);

  // 4) edge phase (v_p now dead -> its buffer becomes agg)
  float* agg = vp_agg;
  init_bufs<<<2048, 256, 0, stream>>>(mbuf, (long)Na * 2, sbuf, (long)Na * 2,
                                      agg, (long)Na * 256);
  {
    int blocks = (int)(((long)E * 32 + 255) / 256);
    edge_logits<<<blocks, 256, 0, stream>>>(eb_src, eb_dst, xp_qa /*q_a*/, ke_za /*ke*/,
                                            p_b, logits, mbuf, E);
    long eh = (long)E * 2;
    edge_expsum<<<(int)((eh + 255) / 256), 256, 0, stream>>>(eb_dst, logits, mbuf, sbuf, eh);
    edge_scatter<<<blocks, 256, 0, stream>>>(eb_src, eb_dst, logits, sbuf, vebuf, agg, E);
  }

  // 5) za = g*(gelu(agg)@Wout_a + bout_a) + (1-g)*xa   (ke dead -> za buffer)
  float* za = ke_za;
  launch_gemm(stream, agg, 256, nullptr, 1 /*gelu on A*/, Wout_a, 256, 0, bout_a,
              za, 256, Na, 256, 256, 3 /*gate-res*/, skip_a, xa, 256);

  // 6) z_in = za[input_nodes] @ W_dec + b_dec
  launch_gemm(stream, za, 256, input_nodes, 0, W_dec, 256, 0, b_dec,
              zin, 256, NI, 256, 256, 0, nullptr, nullptr, 0);

  // 7) out = sigmoid(z_in @ za^T)   -> d_out [NI, Na]
  launch_gemm(stream, zin, 256, nullptr, 0, za, 256, 1 /*B^T*/, nullptr,
              (float*)d_out, Na, NI, Na, 256, 2 /*sigmoid*/, nullptr, nullptr, 0);
}
